// AttentionKVCache_21019569946912
// MI455X (gfx1250) — matrix operand order repaired and checked by reading, not compiled, not run on hardware
//
#include <hip/hip_runtime.h>
#include <hip/hip_bf16.h>
#include <math.h>

#define DIM   2048
#define BATCH 8
#define QL    128
#define KVL   4096
#define KVT   (KVL + QL)     // 4224
#define MROWS (BATCH * QL)   // 1024

typedef __attribute__((ext_vector_type(16))) __bf16 bf16x16;
typedef __attribute__((ext_vector_type(8)))  float  f32x8;
typedef __attribute__((ext_vector_type(4)))  int    v4i;

// Hardware bf16 convert (v_cvt_pk_bf16_f32 on gfx1250) instead of manual RNE.
__device__ __forceinline__ unsigned short f2bf(float f) {
  __bf16 h = (__bf16)f;
  return __builtin_bit_cast(unsigned short, h);
}
__device__ __forceinline__ unsigned int f2bf2(float lo, float hi) {
  return (unsigned int)f2bf(lo) | ((unsigned int)f2bf(hi) << 16);
}

union Frag {
  bf16x16 v;
  unsigned short s[16];
  uint4 q[2];
};

// A-matrix 16x32 bf16 frag (M rows across lanes, K packed in VGPRs):
// lanes 0-15: K = kb+0..7 (v0..3) and kb+16..23 (v4..7); lanes 16-31: +8 / +24.
__device__ __forceinline__ bf16x16 load_a_bf16(const unsigned short* p0, int ld, int lane) {
  Frag f;
  const unsigned short* p = p0 + (size_t)(lane & 15) * ld + (lane >> 4) * 8;
  f.q[0] = *(const uint4*)p;
  f.q[1] = *(const uint4*)(p + 16);
  return f.v;
}

// B-matrix 32x16 bf16 frag from an N x K row-major source (K-contiguous rows):
// lane n<16 holds column N=n, K=kb+0..7 and kb+16..23; lane n+16 holds K=kb+8..15 and kb+24..31.
// Works for both global and LDS sources (generic pointer -> ds_load_b128 in LDS).
__device__ __forceinline__ bf16x16 load_b_bf16(const unsigned short* p0, int ld, int lane) {
  Frag f;
  const unsigned short* p = p0 + (size_t)(lane & 15) * ld + (lane >> 4) * 8;
  f.q[0] = *(const uint4*)p;
  f.q[1] = *(const uint4*)(p + 16);
  return f.v;
}

__device__ __forceinline__ f32x8 wmma_bf16(bf16x16 a, bf16x16 b, f32x8 c) {
  return __builtin_amdgcn_wmma_f32_16x16x32_bf16(false, a, false, b, (short)0, c,
                                                 false, false);
}

// Async global->LDS staging (16B per lane), with sync fallback.
__device__ __forceinline__ void stage_b128(const unsigned short* gsrc,
                                           unsigned short* ldst) {
#if __has_builtin(__builtin_amdgcn_global_load_async_to_lds_b128)
  __builtin_amdgcn_global_load_async_to_lds_b128(
      (__attribute__((address_space(1))) v4i*)gsrc,
      (__attribute__((address_space(3))) v4i*)ldst, 0, 0);
#else
  *(uint4*)ldst = *(const uint4*)gsrc;
#endif
}
__device__ __forceinline__ void stage_wait() {
#if __has_builtin(__builtin_amdgcn_global_load_async_to_lds_b128) && \
    __has_builtin(__builtin_amdgcn_s_wait_asynccnt)
  __builtin_amdgcn_s_wait_asynccnt(0);
#endif
}

// Vectorized f32 -> bf16 conversion (4 elements per thread).
__global__ void cvt_f32_bf16(const float4* __restrict__ src,
                             uint2* __restrict__ dst, int n4) {
  int i = blockIdx.x * blockDim.x + threadIdx.x;
  if (i < n4) {
    float4 t = src[i];
    dst[i] = make_uint2(f2bf2(t.x, t.y), f2bf2(t.z, t.w));
  }
}

// C[M,N] = A[M,K] * B[N,K]^T, bf16 in, f32 accumulate.
// Block = 128 threads (4 waves); block tile 128(M) x 64(N); wave tile 32 x 64.
// B tile (64x32 bf16 = 4KB) staged once per block into LDS (async when available).
template <bool OUT_BF16>
__global__ __launch_bounds__(128) void gemm_abT(
    const unsigned short* __restrict__ A,
    const unsigned short* __restrict__ B,
    void* __restrict__ C, int M, int N, int K) {
  __shared__ unsigned short ldsB[64 * 32];   // [n_local][k_local]
  const int tid = threadIdx.x;
  const int lane = tid & 31;
  const int wv = tid >> 5;
  const int m0 = blockIdx.y * 128 + wv * 32;
  const int n0 = blockIdx.x * 64;
  const unsigned short* Arow = A + (size_t)m0 * K;
  f32x8 acc[2][4] = {};
  for (int k = 0; k < K; k += 32) {
    // Stage B[n0..n0+63][k..k+31]: 256 x 16B chunks, 2 per thread.
    const unsigned short* src = B + (size_t)n0 * K + k;
#pragma unroll
    for (int it = 0; it < 2; ++it) {
      int u4 = tid + it * 128;               // 0..255
      int row = u4 >> 2, c8 = (u4 & 3) * 8;  // 4 chunks per 32-elem row
      stage_b128(src + (size_t)row * K + c8, &ldsB[row * 32 + c8]);
    }
    stage_wait();
    __syncthreads();
    bf16x16 a0 = load_a_bf16(Arow + k, K, lane);
    bf16x16 a1 = load_a_bf16(Arow + (size_t)16 * K + k, K, lane);
#pragma unroll
    for (int t = 0; t < 4; ++t) {
      bf16x16 b = load_b_bf16(&ldsB[t * 16 * 32], 32, lane);
      acc[0][t] = wmma_bf16(a0, b, acc[0][t]);
      acc[1][t] = wmma_bf16(a1, b, acc[1][t]);
    }
    __syncthreads();
  }
  const int r = lane & 15, h = lane >> 4;
#pragma unroll
  for (int u = 0; u < 2; ++u)
#pragma unroll
    for (int t = 0; t < 4; ++t)
#pragma unroll
      for (int i = 0; i < 8; ++i) {
        size_t idx = (size_t)(m0 + u * 16 + h * 8 + i) * N + (size_t)(n0 + t * 16 + r);
        if constexpr (OUT_BF16) ((unsigned short*)C)[idx] = f2bf(acc[u][t][i]);
        else                    ((float*)C)[idx] = acc[u][t][i];
      }
}

// scores[b,q,j] = scale * sum_d qbf[b,q,d] * k[b,j,d]
// Block tile: all 128 q-rows x 64 j; K tile (64 j-rows x 32 d) staged to LDS as
// bf16 once per block (f32 cache converted with hw cvt; bf16 k_new copied).
__global__ __launch_bounds__(128) void scores_kernel(
    const unsigned short* __restrict__ Qb,   // [B*QL, DIM] bf16
    const float* __restrict__ Kc,            // [B, KVL, DIM] f32
    const unsigned short* __restrict__ Kn,   // [B*QL, DIM] bf16
    float* __restrict__ S,                   // [B, QL, KVT] f32
    float scale) {
  __shared__ unsigned short ldsK[64 * 32];   // [j_local][k_local]
  const int b = blockIdx.z;
  const int tid = threadIdx.x;
  const int lane = tid & 31;
  const int wv = tid >> 5;
  const int m0 = wv * 32;                    // 4 waves cover all 128 q rows
  const int n0 = blockIdx.x * 64;            // j tile (uniform vs KVL: 4096 % 64 == 0)
  const bool cachePart = (n0 < KVL);
  const unsigned short* Aq = Qb + ((size_t)b * QL + m0) * DIM;
  const float* srcC = Kc + ((size_t)b * KVL + n0) * DIM;
  const unsigned short* srcN = Kn + ((size_t)b * QL + (n0 >= KVL ? n0 - KVL : 0)) * DIM;
  f32x8 acc[2][4] = {};
  for (int k = 0; k < DIM; k += 32) {
    if (cachePart) {
      // 64 rows x 32 floats = 512 float4 chunks, 4 per thread; hw cvt to bf16.
#pragma unroll
      for (int it = 0; it < 4; ++it) {
        int f4 = tid + it * 128;             // 0..511
        int row = f4 >> 3, c4 = (f4 & 7) * 4;
        float4 t = *(const float4*)(srcC + (size_t)row * DIM + k + c4);
        *(unsigned int*)&ldsK[row * 32 + c4]     = f2bf2(t.x, t.y);
        *(unsigned int*)&ldsK[row * 32 + c4 + 2] = f2bf2(t.z, t.w);
      }
      // Prefetch the next k-slab of the HBM-resident cache stream.
      if (k + 32 < DIM)
        __builtin_prefetch(srcC + (size_t)(tid & 63) * DIM + k + 32, 0, 1);
    } else {
#pragma unroll
      for (int it = 0; it < 2; ++it) {
        int u4 = tid + it * 128;
        int row = u4 >> 2, c8 = (u4 & 3) * 8;
        *(uint4*)&ldsK[row * 32 + c8] = *(const uint4*)(srcN + (size_t)row * DIM + k + c8);
      }
    }
    __syncthreads();
    bf16x16 a0 = load_a_bf16(Aq + k, DIM, lane);
    bf16x16 a1 = load_a_bf16(Aq + (size_t)16 * DIM + k, DIM, lane);
#pragma unroll
    for (int t = 0; t < 4; ++t) {
      bf16x16 bb = load_b_bf16(&ldsK[t * 16 * 32], 32, lane);
      acc[0][t] = wmma_bf16(a0, bb, acc[0][t]);
      acc[1][t] = wmma_bf16(a1, bb, acc[1][t]);
    }
    __syncthreads();
  }
  const int r = lane & 15, h = lane >> 4;
#pragma unroll
  for (int u = 0; u < 2; ++u)
#pragma unroll
    for (int t = 0; t < 4; ++t)
#pragma unroll
      for (int i = 0; i < 8; ++i) {
        int row = m0 + u * 16 + h * 8 + i;
        int col = n0 + t * 16 + r;
        S[((size_t)b * QL + row) * KVT + col] = acc[u][t][i] * scale;
      }
}

// Row softmax over KVT columns; emits bf16 probabilities.
__global__ __launch_bounds__(256) void softmax_kernel(
    const float* __restrict__ S, unsigned short* __restrict__ P) {
  const int row = blockIdx.x;                 // 0 .. B*QL-1
  const float* s = S + (size_t)row * KVT;
  unsigned short* p = P + (size_t)row * KVT;
  __shared__ float red[256];
  const int t = threadIdx.x;

  float m = -3.0e38f;
  for (int c = t; c < KVT; c += 256) m = fmaxf(m, s[c]);
  red[t] = m; __syncthreads();
  for (int st = 128; st > 0; st >>= 1) {
    if (t < st) red[t] = fmaxf(red[t], red[t + st]);
    __syncthreads();
  }
  m = red[0]; __syncthreads();

  float sum = 0.0f;
  for (int c = t; c < KVT; c += 256) sum += __expf(s[c] - m);
  red[t] = sum; __syncthreads();
  for (int st = 128; st > 0; st >>= 1) {
    if (t < st) red[t] += red[t + st];
    __syncthreads();
  }
  const float inv = 1.0f / red[0];

  for (int c = t; c < KVT; c += 256) p[c] = f2bf(__expf(s[c] - m) * inv);
}

// O[b,q,d] = sum_j attn[b,q,j] * v[b,j,d].
// V is K-major (j rows): stage 32(j) x 64(d) tiles through LDS transposed +
// converted, then feed WMMA B frags from LDS. Wave tile 32(q) x 64(d).
__global__ __launch_bounds__(128) void attn_v_kernel(
    const unsigned short* __restrict__ P,    // [B, QL, KVT] bf16
    const float* __restrict__ Vc,            // [B, KVL, DIM] f32
    const unsigned short* __restrict__ Vn,   // [B*QL, DIM] bf16
    unsigned short* __restrict__ O) {        // [B*QL, DIM] bf16
  __shared__ unsigned short ldsB[64 * 32];   // [d_local][j_local]
  const int b = blockIdx.z;
  const int tid = threadIdx.x;
  const int lane = tid & 31;
  const int wv = tid >> 5;
  const int m0 = wv * 32;                    // 4 waves cover all 128 q rows
  const int n0 = blockIdx.x * 64;            // d tile
  const unsigned short* Ap = P + ((size_t)b * QL + m0) * KVT;
  f32x8 acc[2][4] = {};
  for (int k = 0; k < KVT; k += 32) {        // j tile (uniform vs KVL: 4096 % 32 == 0)
    if (k < KVL) {
      const float* src = Vc + ((size_t)b * KVL + k) * DIM + n0;
#pragma unroll
      for (int it = 0; it < 16; ++it) {
        int idx = tid + it * 128;            // 0..2047
        int jl = idx >> 6, dl = idx & 63;
        ldsB[dl * 32 + jl] = f2bf(src[(size_t)jl * DIM + dl]);
      }
      if (k + 32 < KVL)
        __builtin_prefetch(src + (size_t)(tid >> 2) * DIM + 32 * DIM, 0, 1);
    } else {
      const unsigned short* src = Vn + ((size_t)b * QL + (k - KVL)) * DIM + n0;
#pragma unroll
      for (int it = 0; it < 16; ++it) {
        int idx = tid + it * 128;
        int jl = idx >> 6, dl = idx & 63;
        ldsB[dl * 32 + jl] = src[(size_t)jl * DIM + dl];
      }
    }
    __syncthreads();
    bf16x16 a0 = load_a_bf16(Ap + k, KVT, lane);
    bf16x16 a1 = load_a_bf16(Ap + (size_t)16 * KVT + k, KVT, lane);
#pragma unroll
    for (int t = 0; t < 4; ++t) {
      bf16x16 bb = load_b_bf16(&ldsB[t * 16 * 32], 32, lane);
      acc[0][t] = wmma_bf16(a0, bb, acc[0][t]);
      acc[1][t] = wmma_bf16(a1, bb, acc[1][t]);
    }
    __syncthreads();
  }
  const int r = lane & 15, h = lane >> 4;
#pragma unroll
  for (int u = 0; u < 2; ++u)
#pragma unroll
    for (int t = 0; t < 4; ++t)
#pragma unroll
      for (int i = 0; i < 8; ++i) {
        size_t idx = (size_t)(b * QL + m0 + u * 16 + h * 8 + i) * DIM +
                     (size_t)(n0 + t * 16 + r);
        O[idx] = f2bf(acc[u][t][i]);
      }
}

extern "C" void kernel_launch(void* const* d_in, const int* in_sizes, int n_in,
                              void* d_out, int out_size, void* d_ws, size_t ws_size,
                              hipStream_t stream) {
  (void)in_sizes; (void)n_in; (void)out_size; (void)ws_size;
  const float* x  = (const float*)d_in[0];
  // d_in[1] = mask (unused by the reference forward)
  const float* kc = (const float*)d_in[2];
  const float* vc = (const float*)d_in[3];
  const float* wq = (const float*)d_in[4];
  const float* wk = (const float*)d_in[5];
  const float* wv = (const float*)d_in[6];
  const float* wo = (const float*)d_in[7];

  char* cur = (char*)d_ws;
  auto alloc = [&](size_t bytes) -> void* {
    void* p = (void*)cur;
    cur += (bytes + 255) & ~(size_t)255;
    return p;
  };
  unsigned short* x_bf    = (unsigned short*)alloc((size_t)MROWS * DIM * 2);
  unsigned short* wq_bf   = (unsigned short*)alloc((size_t)DIM * DIM * 2);
  unsigned short* wk_bf   = (unsigned short*)alloc((size_t)DIM * DIM * 2);
  unsigned short* wv_bf   = (unsigned short*)alloc((size_t)DIM * DIM * 2);
  unsigned short* wo_bf   = (unsigned short*)alloc((size_t)DIM * DIM * 2);
  unsigned short* q_bf    = (unsigned short*)alloc((size_t)MROWS * DIM * 2);
  unsigned short* kn_bf   = (unsigned short*)alloc((size_t)MROWS * DIM * 2);
  unsigned short* vn_bf   = (unsigned short*)alloc((size_t)MROWS * DIM * 2);
  float*          scores  = (float*)alloc((size_t)BATCH * QL * KVT * 4);
  unsigned short* attn_bf = (unsigned short*)alloc((size_t)BATCH * QL * KVT * 2);
  unsigned short* o_bf    = (unsigned short*)alloc((size_t)MROWS * DIM * 2);

  const int nX4 = MROWS * DIM / 4;
  const int nW4 = DIM * DIM / 4;
  cvt_f32_bf16<<<(nX4 + 255) / 256, 256, 0, stream>>>((const float4*)x,  (uint2*)x_bf,  nX4);
  cvt_f32_bf16<<<(nW4 + 255) / 256, 256, 0, stream>>>((const float4*)wq, (uint2*)wq_bf, nW4);
  cvt_f32_bf16<<<(nW4 + 255) / 256, 256, 0, stream>>>((const float4*)wk, (uint2*)wk_bf, nW4);
  cvt_f32_bf16<<<(nW4 + 255) / 256, 256, 0, stream>>>((const float4*)wv, (uint2*)wv_bf, nW4);
  cvt_f32_bf16<<<(nW4 + 255) / 256, 256, 0, stream>>>((const float4*)wo, (uint2*)wo_bf, nW4);

  dim3 blk(128);
  dim3 gProj(DIM / 64, MROWS / 128);
  gemm_abT<true><<<gProj, blk, 0, stream>>>(x_bf, wq_bf, q_bf,  MROWS, DIM, DIM);
  gemm_abT<true><<<gProj, blk, 0, stream>>>(x_bf, wk_bf, kn_bf, MROWS, DIM, DIM);
  gemm_abT<true><<<gProj, blk, 0, stream>>>(x_bf, wv_bf, vn_bf, MROWS, DIM, DIM);

  const float scale = 1.0f / sqrtf((float)DIM);
  scores_kernel<<<dim3(KVT / 64, 1, BATCH), blk, 0, stream>>>(q_bf, kc, kn_bf, scores, scale);
  softmax_kernel<<<dim3(BATCH * QL), dim3(256), 0, stream>>>(scores, attn_bf);
  attn_v_kernel<<<dim3(DIM / 64, 1, BATCH), blk, 0, stream>>>(attn_bf, vc, vn_bf, o_bf);
  gemm_abT<false><<<gProj, blk, 0, stream>>>(o_bf, wo_bf, d_out, MROWS, DIM, DIM);
}
